// VRNN_30717606101322
// MI455X (gfx1250) — matrix operand order repaired and checked by reading, not compiled, not run on hardware
//
#include <hip/hip_runtime.h>
#include <math.h>

// ---------------------------------------------------------------------------
// VRNN forward for MI455X (gfx1250, wave32, WMMA).
// GEMMs run on v_wmma_f32_16x16x32_bf16. Block = 256 threads (8 waves):
// wave w handles row-half (w&1) of the 16-column stripe (w>>1); 8 accumulator
// tiles per wave (64 VGPRs) -> no scratch spills. K is staged to LDS in
// chunks of 64 (16 WMMAs per barrier pair). BatchNorm stats reduce in
// registers + shfl_xor, then a 1KB LDS combine between row-half waves.
// ---------------------------------------------------------------------------

#define T_STEPS 100
#define BATCH   256
#define XD      64
#define HD      1024
#define ZD      128

typedef __bf16 v16bf __attribute__((ext_vector_type(16)));
typedef float  v8f   __attribute__((ext_vector_type(8)));
typedef unsigned short u16;

__device__ __forceinline__ u16 f2bf(float f) {
  unsigned u = __float_as_uint(f);
  u += 0x7FFFu + ((u >> 16) & 1u);      // round-to-nearest-even
  return (u16)(u >> 16);
}
__device__ __forceinline__ float bf2f(u16 h) {
  return __uint_as_float(((unsigned)h) << 16);
}
__device__ __forceinline__ float softplusf(float x) {
  return (x > 30.f) ? x : log1pf(__expf(x));
}

union FragBf { v16bf v; uint4 q[2]; };   // 32 bytes: 16 x bf16
union Acc    { v8f   v; float f[8]; };   // 16x16 f32 C/D tile slice per lane

#define ACT_NONE     0
#define ACT_RELU     1
#define ACT_SOFTPLUS 2

#define KC     64   // K elements staged per barrier round (2 WMMA-K per tile)
#define APITCH 72   // LDS row pitch in halfs (64 data + 8 pad, conflict-free)

// ---------------------------------------------------------------------------
// GEMM: Y[256,N] = act(BN?(X @ W^T + b)). X is bf16 (optionally split into
// two concatenated sources X0[.,K0] ++ X1[.,strideX1]). W is bf16 [N,K].
// Grid.x = N/64. Block = 256 threads (8 waves):
//   stripe = wave>>1 (16 output cols), mhalf = wave&1 (rows 0-127 / 128-255)
//   per wave: 8 WMMA tiles, 2 K-fragments per staging round.
// ---------------------------------------------------------------------------
__global__ __launch_bounds__(256, 1)
void gemm_wmma(const u16* __restrict__ X0, int K0,
               const u16* __restrict__ X1, int strideX1,
               const u16* __restrict__ W, int K, int N,
               const float* __restrict__ bias,
               const float* __restrict__ gamma,
               const float* __restrict__ beta,
               int act,
               u16* __restrict__ Yb,
               float* __restrict__ Yf)
{
  __shared__ __align__(16) u16 sX[BATCH * APITCH];   // 36 KB activation stage
  __shared__ float sRed[4][2][16][2];                // [stripe][mhalf][col][s,s2]

  const int tid    = threadIdx.x;
  const int lane   = tid & 31;
  const int wave   = tid >> 5;       // 0..7
  const int stripe = wave >> 1;      // 0..3 -> 16-col stripe in 64-col block
  const int mhalf  = wave & 1;       // 0..1 -> rows [0,128) or [128,256)
  const int n      = blockIdx.x * 64 + stripe * 16 + (lane & 15);
  const int half16 = lane >> 4;      // 0 | 1 (lane group within wave)

  Acc acc[8];
#pragma unroll
  for (int t = 0; t < 8; ++t)
#pragma unroll
    for (int r = 0; r < 8; ++r) acc[t].f[r] = 0.f;

  const u16* wRow = W + (size_t)n * K;
  // per-thread row base pointers for staging (hoisted out of the K loop)
  const u16* g0 = X0 + (size_t)tid * K0;
  const u16* g1 = X1 ? (X1 + (size_t)tid * strideX1) : g0;
  uint4* const sRow = (uint4*)(&sX[tid * APITCH]);

  for (int kb = 0; kb < K; kb += KC) {
    // ---- stage 256xKC bf16 chunk of X into LDS (concat-aware) ----
    // Read into temps first so the 8 global b128 loads pipeline, then store.
    {
      const uint4* g = (const uint4*)((!X1 || kb < K0) ? (g0 + kb) : (g1 + (kb - K0)));
      uint4 t0 = g[0], t1 = g[1], t2 = g[2], t3 = g[3];
      uint4 t4 = g[4], t5 = g[5], t6 = g[6], t7 = g[7];
      sRow[0] = t0; sRow[1] = t1; sRow[2] = t2; sRow[3] = t3;
      sRow[4] = t4; sRow[5] = t5; sRow[6] = t6; sRow[7] = t7;
    }
    __syncthreads();

    // ---- two B fragments: 32(K) x 16(N) each; lane L holds col n=L&15,
    //      K = (L&16) + e  -> one contiguous 32B run per lane ----
    FragBf bF0, bF1;
    {
      const uint4* wp = (const uint4*)(wRow + kb + (lane & 16));
      bF0.q[0] = wp[0]; bF0.q[1] = wp[1];
      const uint4* wq = (const uint4*)(wRow + kb + 32 + (lane & 16));
      bF1.q[0] = wq[0]; bF1.q[1] = wq[1];
    }
    if (kb + KC < K) __builtin_prefetch(wRow + kb + KC, 0, 0);  // global_prefetch

    // ---- 8 M-tiles x 2 K-fragments for this wave's row-half ----
#pragma unroll
    for (int t = 0; t < 8; ++t) {
      const u16* ar = &sX[(mhalf * 128 + t * 16 + (lane & 15)) * APITCH + half16 * 16];
      FragBf aF0, aF1;
      aF0.q[0] = *(const uint4*)(ar);
      aF0.q[1] = *(const uint4*)(ar + 8);
      aF1.q[0] = *(const uint4*)(ar + 32);
      aF1.q[1] = *(const uint4*)(ar + 40);
      acc[t].v = __builtin_amdgcn_wmma_f32_16x16x32_bf16(
          false, aF0.v, false, bF0.v, (short)0, acc[t].v, false, false);
      acc[t].v = __builtin_amdgcn_wmma_f32_16x16x32_bf16(
          false, aF1.v, false, bF1.v, (short)0, acc[t].v, false, false);
    }
    __syncthreads();
  }

  // ---- epilogue: bias (+ optional BatchNorm over the 256-row batch) + act ----
  const float bv = bias ? bias[n] : 0.f;
  float scale = 1.f, shift = bv;
  if (gamma) {
    float s = 0.f, s2 = 0.f;
#pragma unroll
    for (int t = 0; t < 8; ++t)
#pragma unroll
      for (int r = 0; r < 8; ++r) {
        const float v = acc[t].f[r] + bv;
        s += v; s2 += v * v;
      }
    // merge the two lane halves (lane L and L^16 share column n): 128 rows
    s  += __shfl_xor(s, 16);
    s2 += __shfl_xor(s2, 16);
    if (lane < 16) { sRed[stripe][mhalf][lane][0] = s; sRed[stripe][mhalf][lane][1] = s2; }
    __syncthreads();
    const float st  = sRed[stripe][0][lane & 15][0] + sRed[stripe][1][lane & 15][0];
    const float s2t = sRed[stripe][0][lane & 15][1] + sRed[stripe][1][lane & 15][1];
    const float mean = st  * (1.f / 256.f);
    const float var  = s2t * (1.f / 256.f) - mean * mean;   // biased var
    const float inv  = rsqrtf(var + 1e-5f);
    scale = gamma[n] * inv;
    shift = beta[n] + (bv - mean) * scale;
  }

#pragma unroll
  for (int t = 0; t < 8; ++t) {
#pragma unroll
    for (int r = 0; r < 8; ++r) {
      const int m = mhalf * 128 + t * 16 + r + half16 * 8;
      float v = acc[t].f[r] * scale + shift;
      if      (act == ACT_RELU)     v = fmaxf(v, 0.f);
      else if (act == ACT_SOFTPLUS) v = softplusf(v);
      if (Yb) Yb[(size_t)m * N + n] = f2bf(v);
      if (Yf) Yf[(size_t)m * N + n] = v;
    }
  }
}

// ---------------------------------------------------------------------------
// Elementwise / reduction kernels
// ---------------------------------------------------------------------------
__global__ void k_convert_bf16(const float* __restrict__ s, u16* __restrict__ d, int n) {
  const int i = blockIdx.x * 256 + threadIdx.x;
  if (i < n) d[i] = f2bf(s[i]);
}

__global__ void k_zero_bf16(u16* __restrict__ p, int n) {
  const int i = blockIdx.x * 256 + threadIdx.x;
  if (i < n) p[i] = 0;
}

__global__ void k_zero2(float* __restrict__ p) {
  if (threadIdx.x == 0 && blockIdx.x == 0) { p[0] = 0.f; p[1] = 0.f; }
}

__global__ void k_sample_z(const float* __restrict__ em, const float* __restrict__ es,
                           const float* __restrict__ ep, u16* __restrict__ z, int n) {
  const int i = blockIdx.x * 256 + threadIdx.x;
  if (i < n) z[i] = f2bf(em[i] + es[i] * ep[i]);
}

__global__ void k_gru(const float* __restrict__ gi, const float* __restrict__ gh,
                      u16* __restrict__ h, int n) {
  const int i = blockIdx.x * 256 + threadIdx.x;
  if (i >= n) return;
  const int b = i >> 10, c = i & 1023;
  const float* gib = gi + (size_t)b * 3072;
  const float* ghb = gh + (size_t)b * 3072;
  const float r  = 1.f / (1.f + __expf(-(gib[c]        + ghb[c])));
  const float z  = 1.f / (1.f + __expf(-(gib[1024 + c] + ghb[1024 + c])));
  const float nn = tanhf(gib[2048 + c] + r * ghb[2048 + c]);
  const float hp = bf2f(h[i]);
  h[i] = f2bf((1.f - z) * nn + z * hp);
}

__global__ void k_kld_mse_partial(const float* __restrict__ em, const float* __restrict__ es,
                                  const float* __restrict__ pm, const float* __restrict__ ps,
                                  const float* __restrict__ dm, const float* __restrict__ x,
                                  float* __restrict__ red) {
  __shared__ float sk[8], sm[8];
  const int i = blockIdx.x * 256 + threadIdx.x;   // 128 blocks cover 32768
  float kld = 0.f, mse = 0.f;
  if (i < BATCH * ZD) {
    const float m1 = em[i], s1 = es[i], m2 = pm[i], s2v = ps[i];
    const float d = m1 - m2;
    kld = 0.5f * (2.f * __logf(s2v) - 2.f * __logf(s1)
                  + (s1 * s1 + d * d) / (s2v * s2v) - 1.f);
  }
  if (i < BATCH * XD) { const float d = dm[i] - x[i]; mse = d * d; }
#pragma unroll
  for (int o = 16; o > 0; o >>= 1) { kld += __shfl_down(kld, o); mse += __shfl_down(mse, o); }
  if ((threadIdx.x & 31) == 0) { sk[threadIdx.x >> 5] = kld; sm[threadIdx.x >> 5] = mse; }
  __syncthreads();
  if (threadIdx.x == 0) {
    float a = 0.f, b = 0.f;
#pragma unroll
    for (int j = 0; j < 8; ++j) { a += sk[j]; b += sm[j]; }
    red[blockIdx.x * 2] = a; red[blockIdx.x * 2 + 1] = b;
  }
}

__global__ void k_reduce_add(const float* __restrict__ red, float* __restrict__ out01, int nb) {
  if (threadIdx.x == 0 && blockIdx.x == 0) {   // deterministic serial add
    float a = 0.f, b = 0.f;
    for (int j = 0; j < nb; ++j) { a += red[2 * j]; b += red[2 * j + 1]; }
    out01[0] += a; out01[1] += b;
  }
}

// ---------------------------------------------------------------------------
// Host orchestration
// ---------------------------------------------------------------------------
extern "C" void kernel_launch(void* const* d_in, const int* in_sizes, int n_in,
                              void* d_out, int out_size, void* d_ws, size_t ws_size,
                              hipStream_t stream) {
  (void)in_sizes; (void)n_in; (void)out_size; (void)ws_size;

  // ---- inputs in pytree-leaf order (top: features, eps, params; nested dicts
  //      sorted by key: lin_bn -> W,b,be,g ; lin -> W,b ; gru -> W_hh,W_ih) ----
  const float* features = (const float*)d_in[0];
  const float* eps      = (const float*)d_in[1];
  int idx = 2;
  auto nf = [&]() { return (const float*)d_in[idx++]; };

  const float *dec_W[4], *dec_b[4], *dec_be[4], *dec_g[4];
  for (int l = 0; l < 4; ++l) { dec_W[l]=nf(); dec_b[l]=nf(); dec_be[l]=nf(); dec_g[l]=nf(); }
  const float *decm_W=nf(), *decm_b=nf();
  const float *decs_W=nf(), *decs_b=nf();
  const float *enc_W[4], *enc_b[4], *enc_be[4], *enc_g[4];
  for (int l = 0; l < 4; ++l) { enc_W[l]=nf(); enc_b[l]=nf(); enc_be[l]=nf(); enc_g[l]=nf(); }
  const float *encm_W=nf(), *encm_b=nf();
  const float *encs_W=nf(), *encs_b=nf();
  const float *gru_Whh=nf(), *gru_Wih=nf();
  const float *px_W[4], *px_b[4], *px_be[4], *px_g[4];
  for (int l = 0; l < 4; ++l) { px_W[l]=nf(); px_b[l]=nf(); px_be[l]=nf(); px_g[l]=nf(); }
  const float *pz_W[3], *pz_b[3], *pz_be[3], *pz_g[3];
  for (int l = 0; l < 3; ++l) { pz_W[l]=nf(); pz_b[l]=nf(); pz_be[l]=nf(); pz_g[l]=nf(); }
  const float *pr_W[3], *pr_b[3];
  for (int l = 0; l < 3; ++l) { pr_W[l]=nf(); pr_b[l]=nf(); }
  const float *prm_W=nf(), *prm_b=nf();
  const float *prs_W=nf(), *prs_b=nf();

  // ---- workspace bump allocator ----
  char* wsp = (char*)d_ws;
  size_t off = 0;
  auto abf  = [&](size_t n)->u16*   { u16* p   = (u16*)  (wsp + off); off += ((n*2 + 255) & ~(size_t)255); return p; };
  auto af32 = [&](size_t n)->float* { float* p = (float*)(wsp + off); off += ((n*4 + 255) & ~(size_t)255); return p; };
  auto conv = [&](const float* s, u16* d, size_t n) {
    k_convert_bf16<<<(unsigned)((n + 255) / 256), 256, 0, stream>>>(s, d, (int)n);
  };

  // ---- weights -> bf16 (L2-resident: ~59 MB total, fits in 192 MB L2) ----
  u16 *bpx[4], *benc[4], *bpr[3], *bpz[3], *bdec[4];
  bpx[0]  = abf((size_t)HD*XD);      conv(px_W[0],  bpx[0],  (size_t)HD*XD);
  for (int l = 1; l < 4; ++l) { bpx[l]  = abf((size_t)HD*HD);   conv(px_W[l],  bpx[l],  (size_t)HD*HD); }
  benc[0] = abf((size_t)HD*2*HD);    conv(enc_W[0], benc[0], (size_t)HD*2*HD);
  for (int l = 1; l < 4; ++l) { benc[l] = abf((size_t)HD*HD);   conv(enc_W[l], benc[l], (size_t)HD*HD); }
  u16* bencm = abf((size_t)ZD*HD);   conv(encm_W, bencm, (size_t)ZD*HD);
  u16* bencs = abf((size_t)ZD*HD);   conv(encs_W, bencs, (size_t)ZD*HD);
  for (int l = 0; l < 3; ++l) { bpr[l]  = abf((size_t)HD*HD);   conv(pr_W[l],  bpr[l],  (size_t)HD*HD); }
  u16* bprm  = abf((size_t)ZD*HD);   conv(prm_W, bprm, (size_t)ZD*HD);
  u16* bprs  = abf((size_t)ZD*HD);   conv(prs_W, bprs, (size_t)ZD*HD);
  bpz[0]  = abf((size_t)HD*ZD);      conv(pz_W[0],  bpz[0],  (size_t)HD*ZD);
  for (int l = 1; l < 3; ++l) { bpz[l]  = abf((size_t)HD*HD);   conv(pz_W[l],  bpz[l],  (size_t)HD*HD); }
  bdec[0] = abf((size_t)HD*2*HD);    conv(dec_W[0], bdec[0], (size_t)HD*2*HD);
  for (int l = 1; l < 4; ++l) { bdec[l] = abf((size_t)HD*HD);   conv(dec_W[l], bdec[l], (size_t)HD*HD); }
  u16* bdecm = abf((size_t)XD*HD);   conv(decm_W, bdecm, (size_t)XD*HD);
  u16* bdecs = abf((size_t)XD*HD);   conv(decs_W, bdecs, (size_t)XD*HD);
  u16* bgih  = abf((size_t)3*HD*2*HD); conv(gru_Wih, bgih, (size_t)3*HD*2*HD);
  u16* bghh  = abf((size_t)3*HD*HD);   conv(gru_Whh, bghh, (size_t)3*HD*HD);

  // ---- activations ----
  u16* xb    = abf((size_t)BATCH*XD);
  u16* bufA  = abf((size_t)BATCH*HD);
  u16* bufB  = abf((size_t)BATCH*HD);
  u16* phix  = abf((size_t)BATCH*HD);
  u16* phiz  = abf((size_t)BATCH*HD);
  u16* hbuf  = abf((size_t)BATCH*HD);
  u16* zb    = abf((size_t)BATCH*ZD);
  float* prm_f = af32((size_t)BATCH*ZD);
  float* prs_f = af32((size_t)BATCH*ZD);
  float* gi_f  = af32((size_t)BATCH*3*HD);
  float* gh_f  = af32((size_t)BATCH*3*HD);
  float* red   = af32(256);

  // ---- output layout: [kld, mse, enc_means, enc_stds, dec_means, dec_stds] ----
  float* out  = (float*)d_out;
  float* o_em = out + 2;
  float* o_es = o_em + (size_t)T_STEPS*BATCH*ZD;
  float* o_dm = o_es + (size_t)T_STEPS*BATCH*ZD;
  float* o_ds = o_dm + (size_t)T_STEPS*BATCH*XD;

  k_zero_bf16<<<(BATCH*HD + 255)/256, 256, 0, stream>>>(hbuf, BATCH*HD);
  k_zero2<<<1, 32, 0, stream>>>(out);

  auto gemm = [&](const u16* X0, int K0, const u16* X1, int sX1,
                  const u16* W, int K, int N,
                  const float* b, const float* g, const float* be,
                  int act, u16* Yb, float* Yf) {
    gemm_wmma<<<N / 64, 256, 0, stream>>>(X0, K0, X1, sX1, W, K, N, b, g, be, act, Yb, Yf);
  };

  for (int t = 0; t < T_STEPS; ++t) {
    const float* x_t   = features + (size_t)t*BATCH*XD;
    const float* eps_t = eps      + (size_t)t*BATCH*ZD;
    float* em_t = o_em + (size_t)t*BATCH*ZD;
    float* es_t = o_es + (size_t)t*BATCH*ZD;
    float* dm_t = o_dm + (size_t)t*BATCH*XD;
    float* ds_t = o_ds + (size_t)t*BATCH*XD;

    k_convert_bf16<<<(BATCH*XD + 255)/256, 256, 0, stream>>>(x_t, xb, BATCH*XD);

    // phi_x: 4 x (Linear + BN + ReLU)
    gemm(xb,   XD, 0,0, bpx[0], XD, HD, px_b[0], px_g[0], px_be[0], ACT_RELU, bufA, 0);
    gemm(bufA, HD, 0,0, bpx[1], HD, HD, px_b[1], px_g[1], px_be[1], ACT_RELU, bufB, 0);
    gemm(bufB, HD, 0,0, bpx[2], HD, HD, px_b[2], px_g[2], px_be[2], ACT_RELU, bufA, 0);
    gemm(bufA, HD, 0,0, bpx[3], HD, HD, px_b[3], px_g[3], px_be[3], ACT_RELU, phix, 0);

    // enc: input = concat(phi_x, h)
    gemm(phix, HD, hbuf, HD, benc[0], 2*HD, HD, enc_b[0], enc_g[0], enc_be[0], ACT_RELU, bufA, 0);
    gemm(bufA, HD, 0,0,      benc[1], HD,   HD, enc_b[1], enc_g[1], enc_be[1], ACT_RELU, bufB, 0);
    gemm(bufB, HD, 0,0,      benc[2], HD,   HD, enc_b[2], enc_g[2], enc_be[2], ACT_RELU, bufA, 0);
    gemm(bufA, HD, 0,0,      benc[3], HD,   HD, enc_b[3], enc_g[3], enc_be[3], ACT_RELU, bufB, 0);
    gemm(bufB, HD, 0,0, bencm, HD, ZD, encm_b, 0,0, ACT_NONE,     0, em_t);
    gemm(bufB, HD, 0,0, bencs, HD, ZD, encs_b, 0,0, ACT_SOFTPLUS, 0, es_t);

    // prior: 3 x (Linear + ReLU), heads
    gemm(hbuf, HD, 0,0, bpr[0], HD, HD, pr_b[0], 0,0, ACT_RELU, bufA, 0);
    gemm(bufA, HD, 0,0, bpr[1], HD, HD, pr_b[1], 0,0, ACT_RELU, bufB, 0);
    gemm(bufB, HD, 0,0, bpr[2], HD, HD, pr_b[2], 0,0, ACT_RELU, bufA, 0);
    gemm(bufA, HD, 0,0, bprm, HD, ZD, prm_b, 0,0, ACT_NONE,     0, prm_f);
    gemm(bufA, HD, 0,0, bprs, HD, ZD, prs_b, 0,0, ACT_SOFTPLUS, 0, prs_f);

    // z = enc_mean + enc_std * eps
    k_sample_z<<<(BATCH*ZD + 255)/256, 256, 0, stream>>>(em_t, es_t, eps_t, zb, BATCH*ZD);

    // phi_z: 3 x (Linear + BN + ReLU)
    gemm(zb,   ZD, 0,0, bpz[0], ZD, HD, pz_b[0], pz_g[0], pz_be[0], ACT_RELU, bufB, 0);
    gemm(bufB, HD, 0,0, bpz[1], HD, HD, pz_b[1], pz_g[1], pz_be[1], ACT_RELU, bufA, 0);
    gemm(bufA, HD, 0,0, bpz[2], HD, HD, pz_b[2], pz_g[2], pz_be[2], ACT_RELU, phiz, 0);

    // dec: input = concat(phi_z, h)
    gemm(phiz, HD, hbuf, HD, bdec[0], 2*HD, HD, dec_b[0], dec_g[0], dec_be[0], ACT_RELU, bufA, 0);
    gemm(bufA, HD, 0,0,      bdec[1], HD,   HD, dec_b[1], dec_g[1], dec_be[1], ACT_RELU, bufB, 0);
    gemm(bufB, HD, 0,0,      bdec[2], HD,   HD, dec_b[2], dec_g[2], dec_be[2], ACT_RELU, bufA, 0);
    gemm(bufA, HD, 0,0,      bdec[3], HD,   HD, dec_b[3], dec_g[3], dec_be[3], ACT_RELU, bufB, 0);
    gemm(bufB, HD, 0,0, bdecm, HD, XD, decm_b, 0,0, ACT_NONE,     0, dm_t);
    gemm(bufB, HD, 0,0, bdecs, HD, XD, decs_b, 0,0, ACT_SOFTPLUS, 0, ds_t);

    // GRU: gi = [phi_x, phi_z] @ W_ih^T ; gh = h @ W_hh^T ; combine
    gemm(phix, HD, phiz, HD, bgih, 2*HD, 3*HD, 0, 0,0, ACT_NONE, 0, gi_f);
    gemm(hbuf, HD, 0,0,      bghh, HD,   3*HD, 0, 0,0, ACT_NONE, 0, gh_f);
    k_gru<<<(BATCH*HD + 255)/256, 256, 0, stream>>>(gi_f, gh_f, hbuf, BATCH*HD);

    // losses (deterministic two-stage reduction)
    k_kld_mse_partial<<<128, 256, 0, stream>>>(em_t, es_t, prm_f, prs_f, dm_t, x_t, red);
    k_reduce_add<<<1, 32, 0, stream>>>(red, out, 128);
  }
}
